// MultiHeadAttention_27436251086965
// MI455X (gfx1250) — matrix operand order repaired and checked by reading, not compiled, not run on hardware
//
#include <hip/hip_runtime.h>
#include <hip/hip_bf16.h>

#define EMBED 768
#define HEADS 12
#define HDIM  64
#define SEQ   2048
#define BATCH 4
#define MROWS (BATCH * SEQ)   // 8192

typedef __attribute__((ext_vector_type(16))) __bf16 v16bf;
typedef __attribute__((ext_vector_type(8)))  __bf16 v8bf;
typedef __attribute__((ext_vector_type(8)))  float  v8f;

union FragU { v16bf v; v8bf h[2]; };

// A-matrix 16x32 bf16 fragment (row-major source, K contiguous).
// lane L (L<16): row=L, halves 0-7 = K 0..7, halves 8-15 = K 16..23.
// lane L (L>=16): row=L-16, halves 0-7 = K 8..15, halves 8-15 = K 24..31.
__device__ __forceinline__ v16bf load_frag_a(const __bf16* base, int lda, int m0, int k0, int lane) {
  const int row  = m0 + (lane & 15);
  const int koff = (lane & 16) ? 8 : 0;
  const __bf16* p = base + (size_t)row * lda + k0 + koff;
  FragU u;
  u.h[0] = *(const v8bf*)(p);
  u.h[1] = *(const v8bf*)(p + 16);
  return u.v;
}

// B-matrix 32x16 bf16 fragment where source is row-major [N, K] (i.e. B^T in memory):
// lane L (L<16): col N=L, halves 0-7 = K 0..7, halves 8-15 = K 16..23; lane L>=16: col N=L-16, halves 0-7 = K 8..15, halves 8-15 = K 24..31.
// Two 16-byte loads per lane.
__device__ __forceinline__ v16bf load_frag_b(const __bf16* base, int ldb, int n0, int k0, int lane) {
  const int col   = n0 + (lane & 15);
  const int kbase = (lane & 16) ? 8 : 0;
  const __bf16* p = base + (size_t)col * ldb + k0 + kbase;
  FragU u;
  u.h[0] = *(const v8bf*)(p);
  u.h[1] = *(const v8bf*)(p + 16);
  return u.v;
}

__device__ __forceinline__ v8f wmma_bf16(v16bf a, v16bf b, v8f c) {
  return __builtin_amdgcn_wmma_f32_16x16x32_bf16(false, a, false, b, (short)0, c, false, false);
}

__global__ void cvt_f32_bf16(const float* __restrict__ src, __bf16* __restrict__ dst, int n) {
  int i = blockIdx.x * blockDim.x + threadIdx.x;
  if (i < n) dst[i] = (__bf16)src[i];
}

// C[M,N] = A[M,K]_bf16 * B[N,K]^T_bf16 + bias, one 32x64 tile per wave
// (2 A-frags reused over 4 B-frags -> 8 WMMAs per k-step, halves operand
// fetch per WMMA vs a 16x64 tile).
// mode 0: bf16 store into [B,H,S,D]   (Q / K)
// mode 2: bf16 store into [B,H,D,S]   (V transposed for the PV B-fragment)
// mode 3: f32 store row-major [M,N]   (final projection -> d_out)
__global__ void gemm_bf16(const __bf16* __restrict__ A, const __bf16* __restrict__ Bm,
                          const float* __restrict__ bias, void* __restrict__ out, int mode) {
  const int lane = threadIdx.x & 31;
  const int wid  = blockIdx.x * (blockDim.x >> 5) + (threadIdx.x >> 5);
  const int nt64 = EMBED / 64;          // 12
  const int m0 = (wid / nt64) * 32;
  const int n0 = (wid % nt64) * 64;

  v8f acc[2][4] = {};
  for (int k0 = 0; k0 < EMBED; k0 += 32) {
    v16bf a0 = load_frag_a(A, EMBED, m0,      k0, lane);
    v16bf a1 = load_frag_a(A, EMBED, m0 + 16, k0, lane);
#pragma unroll
    for (int t = 0; t < 4; ++t) {
      v16bf b = load_frag_b(Bm, EMBED, n0 + t * 16, k0, lane);
      acc[0][t] = wmma_bf16(a0, b, acc[0][t]);
      acc[1][t] = wmma_bf16(a1, b, acc[1][t]);
    }
  }

  // C layout: VGPR r -> row base + r (lanes 0-15) / base + 8 + r (lanes 16-31), col = lane&15.
  const int colL  = lane & 15;
  const int rbase = (lane & 16) ? 8 : 0;
#pragma unroll
  for (int t = 0; t < 4; ++t) {
    const int col = n0 + t * 16 + colL;
    const float bc = bias[col];
    const int hh = col >> 6;     // head index
    const int dd = col & 63;     // dim within head
#pragma unroll
    for (int g = 0; g < 2; ++g) {
#pragma unroll
      for (int r = 0; r < 8; ++r) {
        const int row = m0 + g * 16 + rbase + r;
        const float v = acc[g][t][r] + bc;
        const int bb = row >> 11;        // row / SEQ
        const int ss = row & (SEQ - 1);
        if (mode == 0) {
          ((__bf16*)out)[(((size_t)(bb * HEADS + hh) * SEQ) + ss) * HDIM + dd] = (__bf16)v;
        } else if (mode == 2) {
          ((__bf16*)out)[(((size_t)(bb * HEADS + hh) * HDIM) + dd) * SEQ + ss] = (__bf16)v;
        } else {
          ((float*)out)[(size_t)row * EMBED + col] = v;
        }
      }
    }
  }
}

// Flash-attention: one wave per (b,h, 16-row q tile). Streams keys 32 at a time.
__global__ void attn_kernel(const __bf16* __restrict__ Q, const __bf16* __restrict__ Kc,
                            const __bf16* __restrict__ Vt, __bf16* __restrict__ ctx) {
  __shared__ float ldsP[8][16 * 32];   // per-wave 2KB P-tile staging
  const int lane = threadIdx.x & 31;
  const int w    = threadIdx.x >> 5;
  const int wid  = blockIdx.x * 8 + w;
  const int qt = wid & 127;            // q tile within (b,h)
  const int bh = wid >> 7;             // 0..47
  const __bf16* Qb = Q  + (size_t)bh * SEQ * HDIM;
  const __bf16* Kb = Kc + (size_t)bh * SEQ * HDIM;
  const __bf16* Vb = Vt + (size_t)bh * HDIM * SEQ;
  const int m0 = qt * 16;

  const v16bf aq0 = load_frag_a(Qb, HDIM, m0, 0,  lane);
  const v16bf aq1 = load_frag_a(Qb, HDIM, m0, 32, lane);

  v8f acc[4] = {};
  float mrun[8], lrun[8];
#pragma unroll
  for (int r = 0; r < 8; ++r) { mrun[r] = -__builtin_inff(); lrun[r] = 0.0f; }

  float* P = ldsP[w];
  const float scale = 0.125f;          // 1/sqrt(64)
  const int colL  = lane & 15;
  const int rbase = (lane & 16) ? 8 : 0;

  for (int j0 = 0; j0 < SEQ; j0 += 32) {
    // energy tile 16x32 = two 16x16 C tiles
    v8f c0 = {}, c1 = {};
    c0 = wmma_bf16(aq0, load_frag_b(Kb, HDIM, j0,      0,  lane), c0);
    c0 = wmma_bf16(aq1, load_frag_b(Kb, HDIM, j0,      32, lane), c0);
    c1 = wmma_bf16(aq0, load_frag_b(Kb, HDIM, j0 + 16, 0,  lane), c1);
    c1 = wmma_bf16(aq1, load_frag_b(Kb, HDIM, j0 + 16, 32, lane), c1);

    // online softmax per row (rows striped across 16-lane halves per VGPR)
#pragma unroll
    for (int r = 0; r < 8; ++r) {
      float e0 = c0[r] * scale;
      float e1 = c1[r] * scale;
      float mx = fmaxf(e0, e1);
#pragma unroll
      for (int off = 1; off < 16; off <<= 1)
        mx = fmaxf(mx, __shfl_xor(mx, off, 32));
      float mnew = fmaxf(mrun[r], mx);
      float corr = __expf(mrun[r] - mnew);   // exp(-inf)=0 handles first step
      float p0 = __expf(e0 - mnew);
      float p1 = __expf(e1 - mnew);
      float ps = p0 + p1;
#pragma unroll
      for (int off = 1; off < 16; off <<= 1)
        ps += __shfl_xor(ps, off, 32);
      lrun[r] = lrun[r] * corr + ps;
      mrun[r] = mnew;
#pragma unroll
      for (int t = 0; t < 4; ++t) acc[t][r] *= corr;
      const int row = rbase + r;
      P[row * 32 + colL]      = p0;
      P[row * 32 + colL + 16] = p1;
    }
    asm volatile("s_wait_dscnt 0" ::: "memory");

    // Re-read P as a 16x32 A-fragment (C layout -> A layout transpose via LDS)
    v16bf pa = {};
    {
      const int rowA = lane & 15;
      const int koff = (lane & 16) ? 8 : 0;
      const float* pp = P + rowA * 32 + koff;
#pragma unroll
      for (int q = 0; q < 8; ++q) {
        pa[q]     = (__bf16)pp[q];
        pa[q + 8] = (__bf16)pp[q + 16];
      }
    }

    // PV: B-frag from transposed V (contiguous in j)
#pragma unroll
    for (int t = 0; t < 4; ++t) {
      v16bf bv = load_frag_b(Vb, SEQ, t * 16, j0, lane);
      acc[t] = wmma_bf16(pa, bv, acc[t]);
    }
  }

  // normalize and write ctx bf16 row-major [MROWS, EMBED]
  const int b = bh / HEADS, h = bh % HEADS;
#pragma unroll
  for (int r = 0; r < 8; ++r) {
    const float inv = 1.0f / lrun[r];
    const int s = m0 + rbase + r;
    const size_t base = (((size_t)b * SEQ) + s) * EMBED + (size_t)h * HDIM;
#pragma unroll
    for (int t = 0; t < 4; ++t)
      ctx[base + t * 16 + colL] = (__bf16)(acc[t][r] * inv);
  }
}

extern "C" void kernel_launch(void* const* d_in, const int* in_sizes, int n_in,
                              void* d_out, int out_size, void* d_ws, size_t ws_size,
                              hipStream_t stream) {
  const float* x  = (const float*)d_in[0];
  const float* Wq = (const float*)d_in[1];
  const float* bq = (const float*)d_in[2];
  const float* Wk = (const float*)d_in[3];
  const float* bk = (const float*)d_in[4];
  const float* Wv = (const float*)d_in[5];
  const float* bv = (const float*)d_in[6];
  const float* Wo = (const float*)d_in[7];
  const float* bo = (const float*)d_in[8];

  size_t off = 0;
  auto take = [&](size_t bytes) -> void* {
    void* p = (char*)d_ws + off;
    off += (bytes + 255) & ~(size_t)255;
    return p;
  };
  const size_t XE = (size_t)MROWS * EMBED;   // 6.29M elems
  const size_t WE = (size_t)EMBED * EMBED;   // 0.59M elems
  __bf16* xb  = (__bf16*)take(XE * 2);
  __bf16* wqb = (__bf16*)take(WE * 2);
  __bf16* wkb = (__bf16*)take(WE * 2);
  __bf16* wvb = (__bf16*)take(WE * 2);
  __bf16* wob = (__bf16*)take(WE * 2);
  __bf16* Qb  = (__bf16*)take(XE * 2);
  __bf16* Kb  = (__bf16*)take(XE * 2);
  __bf16* Vt  = (__bf16*)take(XE * 2);
  __bf16* ctx = (__bf16*)take(XE * 2);

  cvt_f32_bf16<<<(int)((XE + 255) / 256), 256, 0, stream>>>(x,  xb,  (int)XE);
  cvt_f32_bf16<<<(int)((WE + 255) / 256), 256, 0, stream>>>(Wq, wqb, (int)WE);
  cvt_f32_bf16<<<(int)((WE + 255) / 256), 256, 0, stream>>>(Wk, wkb, (int)WE);
  cvt_f32_bf16<<<(int)((WE + 255) / 256), 256, 0, stream>>>(Wv, wvb, (int)WE);
  cvt_f32_bf16<<<(int)((WE + 255) / 256), 256, 0, stream>>>(Wo, wob, (int)WE);

  // 32x64 tile per wave: (8192/32)*(768/64) = 3072 waves -> 384 blocks of 8 waves
  const int gemm_blocks = (MROWS / 32) * (EMBED / 64) / 8;
  gemm_bf16<<<gemm_blocks, 256, 0, stream>>>(xb, wqb, bq, Qb, 0);
  gemm_bf16<<<gemm_blocks, 256, 0, stream>>>(xb, wkb, bk, Kb, 0);
  gemm_bf16<<<gemm_blocks, 256, 0, stream>>>(xb, wvb, bv, Vt, 2);

  const int attn_blocks = (BATCH * HEADS * (SEQ / 16)) / 8; // 768
  attn_kernel<<<attn_blocks, 256, 0, stream>>>(Qb, Kb, Vt, ctx);

  gemm_bf16<<<gemm_blocks, 256, 0, stream>>>(ctx, wob, bo, d_out, 3);
}
